// CrossNeighborAttention_52596169507405
// MI455X (gfx1250) — matrix operand order repaired and checked by reading, not compiled, not run on hardware
//
#include <hip/hip_runtime.h>

// ---------------------------------------------------------------------------
// CrossNeighborAttention for MI455X (gfx1250, wave32)
//   - all matmuls via v_wmma_f32_16x16x32_f16
//   - all LDS staging via global_load_async_to_lds_b128, double-buffered so
//     copies of tile s+1 overlap WMMAs of tile s (s_wait_asynccnt 2/4 pipeline)
// Pipeline:
//   cvt_f16      : img, Wq, Wk, Wv, Wo  f32 -> f16 staging buffers
//   gemm mode 0  : Q = RoPE(X @ Wq^T) * 0.125 -> f16 [b][h][p][64]
//   gemm mode 1  : K = RoPE(X @ Wk^T)         -> f16 [b][h][p][64]
//   gemm mode 2  : V = X @ Wv^T               -> f16 [b][h][t][d][n] (transposed)
//   attn         : flash softmax(Q K_ca^T) V_ca -> f16 [b][t][n][C]
//   gemm mode 3  : out = O @ Wo^T + bo        -> f32
// ---------------------------------------------------------------------------

#define Bc 2
#define Tc 8
#define Nc 576
#define Cc 1024
#define Hc 16
#define HDc 64
#define TNc (Tc * Nc)          // 4608
#define Mrows (Bc * TNc)       // 9216

typedef _Float16 v16h __attribute__((ext_vector_type(16)));
typedef float    v8f  __attribute__((ext_vector_type(8)));

union Frag16 { v16h v; _Float16 h[16]; };

// A-matrix (16x32 f16) K index for slot s (0..15) given lane half (lane>=16)
__device__ __forceinline__ int kslotA(int s, int half) {
  return (s < 8) ? (half ? s + 8 : s) : (half ? s + 16 : s + 8);
}
// B-matrix (32x16 f16) K index for slot s (0..15): same order as kslotA
__device__ __forceinline__ int kslotB(int s, int half) {
  return (s < 8) ? (half ? s + 8 : s) : (half ? s + 16 : s + 8);
}

// Async 16-byte global -> LDS copy (per-lane addresses), tracked by ASYNCcnt.
__device__ __forceinline__ void async_ld_b128(unsigned lds_byte_addr, const void* g) {
  unsigned long long ga = (unsigned long long)(uintptr_t)g;
  asm volatile("global_load_async_to_lds_b128 %0, %1, off"
               :: "v"(lds_byte_addr), "v"(ga) : "memory");
}
// ASYNC loads complete in order -> waiting "<= N" retires all older copies.
__device__ __forceinline__ void wait_async_0() {
  asm volatile("s_wait_asynccnt 0x0" ::: "memory");
}
__device__ __forceinline__ void wait_async_2() {
  asm volatile("s_wait_asynccnt 0x2" ::: "memory");
}
__device__ __forceinline__ void wait_async_4() {
  asm volatile("s_wait_asynccnt 0x4" ::: "memory");
}

// ---------------------------------------------------------------------------
__global__ __launch_bounds__(256) void cvt_f16(const float* __restrict__ in,
                                               _Float16* __restrict__ out, int n) {
  int i = blockIdx.x * 256 + threadIdx.x;
  int stride = gridDim.x * 256;
  for (; i < n; i += stride) out[i] = (_Float16)in[i];
}

// ---------------------------------------------------------------------------
// Tiled GEMM: OUT(M x 1024) = X(M x 1024) @ W(1024 x 1024)^T, 64x64 tile/block.
// Dynamic LDS: As0|Bs0|As1|Bs1 (4KB each, double buffer) | Cs f32[64][64]
// ---------------------------------------------------------------------------
#define G_AS0 0u
#define G_BS0 4096u
#define G_AS1 8192u
#define G_BS1 12288u
#define G_CS  16384u
#define G_SMEM (16384u + 16384u)

__global__ __launch_bounds__(256) void gemm_kernel(
    const _Float16* __restrict__ X, const _Float16* __restrict__ W,
    const float* __restrict__ fcos, const float* __restrict__ fsin,
    const float* __restrict__ bias,
    _Float16* __restrict__ dst16, float* __restrict__ dst32, int mode)
{
  extern __shared__ char smem[];
  float* Cs = (float*)(smem + G_CS);
  const unsigned lbase = (unsigned)__builtin_amdgcn_groupstaticsize();
  const unsigned asOff[2] = {G_AS0, G_AS1};
  const unsigned bsOff[2] = {G_BS0, G_BS1};

  const int m0   = blockIdx.x * 64;
  const int n0   = blockIdx.y * 64;
  const int tid  = threadIdx.x;
  const int w    = tid >> 5;
  const int lane = tid & 31;
  const int half = lane >> 4;
  const int lrow = lane & 15;
  const int ti   = w >> 1;          // row tile 0..3
  const int tj0  = (w & 1) * 2;     // first of two col tiles

  // this thread's 16-byte async chunk within a 64x32 f16 tile
  const int ar = tid >> 2;          // row 0..63
  const int ac = (tid & 3) * 8;     // col 0..24 step 8
  const unsigned chunkOff = (unsigned)(ar * 32 + ac) * 2;
  const _Float16* Xrow = X + (size_t)(m0 + ar) * Cc + ac;
  const _Float16* Wrow = W + (size_t)(n0 + ar) * Cc + ac;

  v8f acc0 = {}; v8f acc1 = {};

  // prologue: stage 0 -> buffer 0
  async_ld_b128(lbase + G_AS0 + chunkOff, Xrow);
  async_ld_b128(lbase + G_BS0 + chunkOff, Wrow);

  for (int s = 0; s < Cc / 32; ++s) {
    if (s + 1 < Cc / 32) {          // prefetch stage s+1 into the other buffer
      unsigned d = (unsigned)((s + 1) & 1);
      async_ld_b128(lbase + asOff[d] + chunkOff, Xrow + (s + 1) * 32);
      async_ld_b128(lbase + bsOff[d] + chunkOff, Wrow + (s + 1) * 32);
      wait_async_2();               // retire stage s copies (in-order)
    } else {
      wait_async_0();
    }
    __syncthreads();

    const _Float16* As = (const _Float16*)(smem + asOff[s & 1]);
    const _Float16* Bs = (const _Float16*)(smem + bsOff[s & 1]);
    Frag16 a, b0, b1;
    #pragma unroll
    for (int k = 0; k < 16; ++k) a.h[k] = As[(ti * 16 + lrow) * 32 + kslotA(k, half)];
    #pragma unroll
    for (int k = 0; k < 16; ++k) {
      int kb = kslotB(k, half);
      b0.h[k] = Bs[(tj0 * 16 + lrow) * 32 + kb];
      b1.h[k] = Bs[((tj0 + 1) * 16 + lrow) * 32 + kb];
    }
    acc0 = __builtin_amdgcn_wmma_f32_16x16x32_f16(false, a.v, false, b0.v,
                                                  (short)0, acc0, false, false);
    acc1 = __builtin_amdgcn_wmma_f32_16x16x32_f16(false, a.v, false, b1.v,
                                                  (short)0, acc1, false, false);
    __syncthreads();
  }

  #pragma unroll
  for (int r = 0; r < 8; ++r) {
    int row = ti * 16 + r + 8 * half;
    Cs[row * 64 + tj0 * 16 + lrow]       = acc0[r];
    Cs[row * 64 + (tj0 + 1) * 16 + lrow] = acc1[r];
  }
  __syncthreads();

  if (mode == 3) {                      // final projection: f32 + bias
    for (int e = tid; e < 64 * 64; e += 256) {
      int row = e >> 6, col = e & 63;
      int gi = m0 + row, gj = n0 + col;
      dst32[(size_t)gi * Cc + gj] = Cs[row * 64 + col] + bias[gj];
    }
  } else if (mode == 2) {               // V: transposed head layout [b][h][t][d][n]
    for (int e = tid; e < 64 * 64; e += 256) {
      int row = e >> 6, col = e & 63;
      int gi = m0 + row, gj = n0 + col;
      int h = gj >> 6, d = gj & 63;
      int bb = gi / TNc, p = gi - bb * TNc;
      int tt = p / Nc, nn = p - tt * Nc;
      size_t di = ((((size_t)(bb * Hc + h) * Tc + tt) * HDc + d) * Nc) + nn;
      dst16[di] = (_Float16)Cs[row * 64 + col];
    }
  } else {                              // Q/K: RoPE (+0.125 scale for Q)
    const float sc = (mode == 0) ? 0.125f : 1.0f;
    for (int pe = tid; pe < 64 * 32; pe += 256) {
      int row = pe >> 5, pc = pe & 31;
      int col0 = pc * 2;
      int gi = m0 + row, gj = n0 + col0;
      int h = gj >> 6, d0 = gj & 63;
      int bb = gi / TNc, p = gi - bb * TNc;
      float c0 = fcos[(size_t)p * HDc + d0];
      float s0 = fsin[(size_t)p * HDc + d0];
      float c1 = fcos[(size_t)p * HDc + d0 + 1];
      float s1 = fsin[(size_t)p * HDc + d0 + 1];
      float x0 = Cs[row * 64 + col0], x1 = Cs[row * 64 + col0 + 1];
      float y0 = (x0 * c0 - x1 * s0) * sc;
      float y1 = (x1 * c1 + x0 * s1) * sc;
      size_t di = ((size_t)(bb * Hc + h) * TNc + p) * HDc + d0;
      dst16[di]     = (_Float16)y0;
      dst16[di + 1] = (_Float16)y1;
    }
  }
}

// ---------------------------------------------------------------------------
// Flash-style neighbor attention. One block = (b, h, t, 64-row n-tile).
// Q lives in registers (A-fragments, loaded once). K/V chunks double-buffered
// via async-to-LDS. Dynamic LDS:
//   K0|V0|K1|V1 (8KB each) | Ss f32[64][64] | Ps f16[64][64] | rm|rl|rs
// ---------------------------------------------------------------------------
#define A_K0 0u
#define A_V0 8192u
#define A_K1 16384u
#define A_V1 24576u
#define A_SS 32768u
#define A_PS 49152u
#define A_RM 57344u
#define A_RL 57600u
#define A_RS 57856u
#define A_SMEM 58112u

__global__ __launch_bounds__(256) void attn_kernel(
    const _Float16* __restrict__ Qb, const _Float16* __restrict__ Kb,
    const _Float16* __restrict__ Vb, _Float16* __restrict__ Ob)
{
  extern __shared__ char smem[];
  float*    Ss = (float*)(smem + A_SS);
  _Float16* Ps = (_Float16*)(smem + A_PS);
  float*    rm = (float*)(smem + A_RM);
  float*    rl = (float*)(smem + A_RL);
  float*    rs = (float*)(smem + A_RS);
  const unsigned lbase = (unsigned)__builtin_amdgcn_groupstaticsize();
  const unsigned kOff[2] = {A_K0, A_K1};
  const unsigned vOff[2] = {A_V0, A_V1};

  int bid = blockIdx.x;
  const int nt = bid % (Nc / 64);  bid /= (Nc / 64);
  const int t  = bid % Tc;         bid /= Tc;
  const int h  = bid % Hc;
  const int b  = bid / Hc;

  const int tid  = threadIdx.x;
  const int w    = tid >> 5;
  const int lane = tid & 31;
  const int half = lane >> 4;
  const int lrow = lane & 15;
  const int ti   = w >> 1;
  const int tj0  = (w & 1) * 2;

  const size_t headBase = (size_t)(b * Hc + h) * TNc;   // Q,K: [b][h][p][d]

  const int NCH = (2 * Nc) / 64;                        // 18 m-chunks

  // issue async K/V copies for chunk c into buffers at kDst/vDst
  auto issue_chunk = [&](int c, unsigned kDst, unsigned vDst) {
    int m0 = c * 64, tsrc, ms;
    if (m0 < Nc) { tsrc = (t == 0) ? 1 : t - 1;           ms = m0;      }
    else         { tsrc = (t == Tc - 1) ? Tc - 2 : t + 1; ms = m0 - Nc; }
    const _Float16* Kp = Kb + (headBase + (size_t)tsrc * Nc + ms) * HDc;      // [m][d]
    const _Float16* Vp = Vb + ((size_t)(b * Hc + h) * Tc + tsrc) * (size_t)HDc * Nc + ms; // [d][n]
    #pragma unroll
    for (int it = 0; it < 2; ++it) {
      int q = tid + it * 256;                 // 512 x 16B chunks per 8KB tile
      int r = q >> 3, cc2 = (q & 7) * 8;
      async_ld_b128(lbase + kDst + (unsigned)(r * 64 + cc2) * 2,
                    Kp + (size_t)r * HDc + cc2);
      async_ld_b128(lbase + vDst + (unsigned)(r * 64 + cc2) * 2,
                    Vp + (size_t)r * Nc + cc2);
    }
  };

  // prologue: chunk 0 -> buffer 0
  issue_chunk(0, A_K0, A_V0);

  // Q A-fragments straight from global into registers (this wave's 16 rows)
  Frag16 qa[2];
  {
    const int p0 = t * Nc + nt * 64;
    const size_t qrow = (headBase + p0 + ti * 16 + lrow) * (size_t)HDc;
    #pragma unroll
    for (int ksi = 0; ksi < 2; ++ksi)
      #pragma unroll
      for (int k = 0; k < 16; ++k)
        qa[ksi].h[k] = Qb[qrow + ksi * 32 + kslotA(k, half)];
  }
  if (tid < 64) { rm[tid] = -3.0e38f; rl[tid] = 0.0f; }

  v8f o0 = {}; v8f o1 = {};

  for (int c = 0; c < NCH; ++c) {
    if (c + 1 < NCH) {                 // prefetch next chunk into other buffer
      issue_chunk(c + 1, kOff[(c + 1) & 1], vOff[(c + 1) & 1]);
      wait_async_4();                  // retire chunk c copies (in-order)
    } else {
      wait_async_0();
    }
    __syncthreads();
    const _Float16* Kt = (const _Float16*)(smem + kOff[c & 1]);   // [m][d]
    const _Float16* Vt = (const _Float16*)(smem + vOff[c & 1]);   // [d][m]

    // S = Q @ K^T for this 64x64 chunk (K-dim = 64 -> two wmma steps)
    v8f sA = {}; v8f sB = {};
    #pragma unroll
    for (int ks = 0; ks < 64; ks += 32) {
      Frag16 b0, b1;
      #pragma unroll
      for (int k = 0; k < 16; ++k) {
        int kb = ks + kslotB(k, half);
        b0.h[k] = Kt[(tj0 * 16 + lrow) * 64 + kb];
        b1.h[k] = Kt[((tj0 + 1) * 16 + lrow) * 64 + kb];
      }
      sA = __builtin_amdgcn_wmma_f32_16x16x32_f16(false, qa[ks >> 5].v, false, b0.v,
                                                  (short)0, sA, false, false);
      sB = __builtin_amdgcn_wmma_f32_16x16x32_f16(false, qa[ks >> 5].v, false, b1.v,
                                                  (short)0, sB, false, false);
    }
    #pragma unroll
    for (int r = 0; r < 8; ++r) {
      int row = ti * 16 + r + 8 * half;
      Ss[row * 64 + tj0 * 16 + lrow]       = sA[r];
      Ss[row * 64 + (tj0 + 1) * 16 + lrow] = sB[r];
    }
    __syncthreads();

    // online softmax row update (1/sqrt(d) already folded into Q)
    if (tid < 64) {
      float mx = rm[tid], cmax = -3.0e38f;
      for (int j = 0; j < 64; ++j) cmax = fmaxf(cmax, Ss[tid * 64 + j]);
      float nm = fmaxf(mx, cmax);
      float scale = __expf(mx - nm);
      float ls = 0.0f;
      for (int j = 0; j < 64; ++j) {
        float pv = __expf(Ss[tid * 64 + j] - nm);
        ls += pv;
        Ps[tid * 64 + j] = (_Float16)pv;
      }
      rs[tid] = scale;
      rm[tid] = nm;
      rl[tid] = rl[tid] * scale + ls;
    }
    __syncthreads();

    // rescale running O, then O += P @ V
    #pragma unroll
    for (int r = 0; r < 8; ++r) {
      float scv = rs[ti * 16 + r + 8 * half];
      o0[r] *= scv;
      o1[r] *= scv;
    }
    #pragma unroll
    for (int ks = 0; ks < 64; ks += 32) {
      Frag16 a, b0, b1;
      #pragma unroll
      for (int k = 0; k < 16; ++k)
        a.h[k] = Ps[(ti * 16 + lrow) * 64 + ks + kslotA(k, half)];
      #pragma unroll
      for (int k = 0; k < 16; ++k) {
        int kb = ks + kslotB(k, half);
        b0.h[k] = Vt[(tj0 * 16 + lrow) * 64 + kb];
        b1.h[k] = Vt[((tj0 + 1) * 16 + lrow) * 64 + kb];
      }
      o0 = __builtin_amdgcn_wmma_f32_16x16x32_f16(false, a.v, false, b0.v,
                                                  (short)0, o0, false, false);
      o1 = __builtin_amdgcn_wmma_f32_16x16x32_f16(false, a.v, false, b1.v,
                                                  (short)0, o1, false, false);
    }
    __syncthreads();
  }

  // normalize by row sum and store to [b][t][n][h*64+d] (f16 GEMM input)
  #pragma unroll
  for (int r = 0; r < 8; ++r) {
    int row = ti * 16 + r + 8 * half;
    float inv = 1.0f / rl[row];
    int nloc = nt * 64 + row;
    size_t oi = (((size_t)b * Tc + t) * Nc + nloc) * Cc + h * HDc;
    Ob[oi + tj0 * 16 + lrow]       = (_Float16)(o0[r] * inv);
    Ob[oi + (tj0 + 1) * 16 + lrow] = (_Float16)(o1[r] * inv);
  }
}

// ---------------------------------------------------------------------------
extern "C" void kernel_launch(void* const* d_in, const int* in_sizes, int n_in,
                              void* d_out, int out_size, void* d_ws, size_t ws_size,
                              hipStream_t stream) {
  const float* img  = (const float*)d_in[0];
  const float* fcos = (const float*)d_in[1];
  const float* fsin = (const float*)d_in[2];
  const float* Wq   = (const float*)d_in[3];
  const float* Wk   = (const float*)d_in[4];
  const float* Wv   = (const float*)d_in[5];
  const float* Wo   = (const float*)d_in[6];
  const float* bo   = (const float*)d_in[7];
  float* out = (float*)d_out;

  const size_t QKV_ELEMS = (size_t)Bc * Hc * TNc * HDc;   // 9,437,184
  const size_t X_ELEMS   = (size_t)Mrows * Cc;            // 9,437,184
  const size_t W_ELEMS   = (size_t)Cc * Cc;               // 1,048,576

  _Float16* Qb  = (_Float16*)d_ws;
  _Float16* Kb  = Qb + QKV_ELEMS;
  _Float16* Vb  = Kb + QKV_ELEMS;           // transposed layout [b][h][t][d][n]
  _Float16* Ob  = Vb + QKV_ELEMS;
  _Float16* Xf  = Ob + QKV_ELEMS;
  _Float16* Wqf = Xf + X_ELEMS;
  _Float16* Wkf = Wqf + W_ELEMS;
  _Float16* Wvf = Wkf + W_ELEMS;
  _Float16* Wof = Wvf + W_ELEMS;

  cvt_f16<<<4096, 256, 0, stream>>>(img, Xf, (int)X_ELEMS);
  cvt_f16<<<1024, 256, 0, stream>>>(Wq, Wqf, (int)W_ELEMS);
  cvt_f16<<<1024, 256, 0, stream>>>(Wk, Wkf, (int)W_ELEMS);
  cvt_f16<<<1024, 256, 0, stream>>>(Wv, Wvf, (int)W_ELEMS);
  cvt_f16<<<1024, 256, 0, stream>>>(Wo, Wof, (int)W_ELEMS);

  dim3 gg(Mrows / 64, Cc / 64);   // 144 x 16 tiles
  gemm_kernel<<<gg, 256, G_SMEM, stream>>>(Xf, Wqf, fcos, fsin, nullptr,
                                           Qb, nullptr, 0);
  gemm_kernel<<<gg, 256, G_SMEM, stream>>>(Xf, Wkf, fcos, fsin, nullptr,
                                           Kb, nullptr, 1);
  gemm_kernel<<<gg, 256, G_SMEM, stream>>>(Xf, Wvf, nullptr, nullptr, nullptr,
                                           Vb, nullptr, 2);

  attn_kernel<<<Bc * Hc * Tc * (Nc / 64), 256, A_SMEM, stream>>>(Qb, Kb, Vb, Ob);

  gemm_kernel<<<gg, 256, G_SMEM, stream>>>(Ob, Wof, nullptr, nullptr, bo,
                                           nullptr, out, 3);
}
